// RelationalTransformerUpdate_27075473834757
// MI455X (gfx1250) — matrix operand order repaired and checked by reading, not compiled, not run on hardware
//
#include <hip/hip_runtime.h>

// ---------------------------------------------------------------------------
// CDNA5 (gfx1250) relational transformer forward.
// All GEMMs via v_wmma_f32_16x16x32_bf16 (wave32). Main GEMM register-blocks
// each wave to a 32x64 output tile: 8 independent WMMA accumulators per
// k-step (hides XDL latency, 8 WMMA : 6 fragment loads).
// ---------------------------------------------------------------------------

typedef __bf16 v16bf __attribute__((ext_vector_type(16)));
typedef float  v8f   __attribute__((ext_vector_type(8)));

#define NL    8
#define BB    8
#define LL    512
#define HH    256
#define NH    8
#define DK    32
#define FFD   1024
#define NREL  37
#define RPAD  40
#define ROWS  (BB * LL)        // 4096
#define BHROW (BB * NH * LL)   // 32768
#define RSQRT_DK    0.17677669529663689f   // 1/sqrt(32)
#define ALIGN_SCALE 0.0625f                // 1/sqrt(256)

#define GF_RELU  1
#define GF_RESID 2
#define GF_BF16  4
#define GF_F32   8
#define GF_QKV   16

// --------------------------- WMMA helpers ----------------------------------

__device__ __forceinline__ v8f wmma_bf16(v16bf a, v16bf b, v8f c) {
  // D = A(16x32) * B(32x16) + C(16x16), f32 accum
  return __builtin_amdgcn_wmma_f32_16x16x32_bf16(false, a, false, b, (short)0, c,
                                                 false, false);
}

// A fragment from row-major [M,K]: lane<16 holds K 0-15; lane>=16 holds
// K 16-31 (same k order as the B fragments).
__device__ __forceinline__ v16bf load_afrag(const __bf16* base, int ld, int m0,
                                            int k0, int lane) {
  const __bf16* p = base + (size_t)(m0 + (lane & 15)) * ld + k0 + ((lane & 16) ? 16 : 0);
  v16bf a;
#pragma unroll
  for (int e = 0; e < 8; ++e) a[e] = p[e];
#pragma unroll
  for (int e = 0; e < 8; ++e) a[8 + e] = p[8 + e];
  return a;
}

// B fragment where the source is stored [N,K] row-major (K contiguous):
// lane gives N = lane&15; lanes 0-15 cover K 0-15, lanes 16-31 cover K 16-31.
__device__ __forceinline__ v16bf load_bfrag_nk(const __bf16* base, int ld, int n0,
                                               int k0, int lane) {
  const __bf16* p = base + (size_t)(n0 + (lane & 15)) * ld + k0 + ((lane & 16) ? 16 : 0);
  v16bf b;
#pragma unroll
  for (int e = 0; e < 16; ++e) b[e] = p[e];
  return b;
}

// B fragment where the source is stored [K,N] row-major (stride ld along K).
__device__ __forceinline__ v16bf load_bfrag_kn(const __bf16* base, int ld, int k0,
                                               int n0, int lane) {
  const __bf16* p = base + (size_t)(k0 + ((lane & 16) ? 16 : 0)) * ld + n0 + (lane & 15);
  v16bf b;
#pragma unroll
  for (int e = 0; e < 16; ++e) b[e] = p[(size_t)e * ld];
  return b;
}

// --------------------------- utility kernels -------------------------------

// f32 [batch][K][N] -> bf16 [batch][N][K] (weights transposed so GEMM B-operand
// loads are K-contiguous).
__global__ void transpose_bf16_k(const float* __restrict__ src, __bf16* __restrict__ dst,
                                 int K, int N, size_t sStride, size_t dStride) {
  int batch = blockIdx.y;
  int t = blockIdx.x * blockDim.x + threadIdx.x;
  if (t >= K * N) return;
  int k = t / N, n = t % N;
  dst[batch * dStride + (size_t)n * K + k] = (__bf16)src[batch * sStride + t];
}

__global__ void qkv_bias_concat(const float* __restrict__ bq, const float* __restrict__ bk,
                                const float* __restrict__ bv, float* __restrict__ dst) {
  int i = blockIdx.x * blockDim.x + threadIdx.x;  // NL*768
  if (i >= NL * 768) return;
  int layer = i / 768, n = i % 768;
  const float* s = (n < 256) ? bq : (n < 512) ? bk : bv;
  dst[i] = s[layer * 256 + (n & 255)];
}

// LayerNorm over 256 cols; one block per row; optional f32 / bf16 outputs.
__global__ void ln_kernel(const float* __restrict__ x, const float* __restrict__ g,
                          const float* __restrict__ b, float* outf, __bf16* outb) {
  __shared__ float red[HH];
  int row = blockIdx.x, t = threadIdx.x;
  float v = x[(size_t)row * HH + t];
  red[t] = v;
  __syncthreads();
  for (int s = HH / 2; s; s >>= 1) { if (t < s) red[t] += red[t + s]; __syncthreads(); }
  float mean = red[0] * (1.0f / HH);
  __syncthreads();
  float d = v - mean;
  red[t] = d * d;
  __syncthreads();
  for (int s = HH / 2; s; s >>= 1) { if (t < s) red[t] += red[t + s]; __syncthreads(); }
  float var = red[0] * (1.0f / HH);
  float y = g[t] * d * rsqrtf(var + 1e-6f) + b[t];
  if (outf) outf[(size_t)row * HH + t] = y;
  if (outb) outb[(size_t)row * HH + t] = (__bf16)y;
}

// qrel[row][r] = scale * dot(q[row,:dk], relk[r,:dk])   (relation bias table)
__global__ void qrel_kernel(const __bf16* __restrict__ q, const float* __restrict__ relk,
                            float* __restrict__ out, int nrows, int dk, int ld, float scale) {
  int idx = blockIdx.x * blockDim.x + threadIdx.x;
  int row = idx / NREL, r = idx % NREL;
  if (row >= nrows) return;
  const __bf16* qp = q + (size_t)row * ld;
  const float* rp = relk + (size_t)r * dk;
  float s = 0.f;
  for (int d = 0; d < dk; ++d) s += (float)qp[d] * rp[d];
  out[(size_t)row * RPAD + r] = s * scale;
}

// ------------------------------- GEMM --------------------------------------
// C[M,N] = A[M,K](bf16) * Bt[N,K]^T(bf16) + bias.  One wave per 32x64 output
// tile: 2 A frags x 4 B frags -> 8 independent WMMA accumulators per k-step.
// Optional relu / residual / bf16 store / QKV head-scatter epilogues.
__global__ void gemm_wmma_k(const __bf16* __restrict__ A, int lda,
                            const __bf16* __restrict__ Bt, int ldb,
                            const float* __restrict__ bias, const float* __restrict__ resid,
                            float* __restrict__ Cf, __bf16* __restrict__ Cb,
                            int M, int N, int K, int flags,
                            __bf16* __restrict__ qd, __bf16* __restrict__ kd,
                            __bf16* __restrict__ vd) {
  int lane = threadIdx.x & 31;
  int wave = threadIdx.x >> 5;
  int tilesN = N >> 6;                       // 64-wide N tiles
  int wt = blockIdx.x * 4 + wave;
  if (wt >= (M >> 5) * tilesN) return;
  int m0 = (wt / tilesN) << 5;               // 32-tall M tiles
  int n0 = (wt % tilesN) << 6;

  v8f c0[4] = {}, c1[4] = {};
  for (int kk = 0; kk < K; kk += 32) {
    if (kk + 32 < K) {
      __builtin_prefetch(A + (size_t)(m0 + (lane & 15)) * lda + kk + 32, 0, 3);
      __builtin_prefetch(Bt + (size_t)(n0 + (lane & 15)) * ldb + kk + 32, 0, 3);
    }
    v16bf a0 = load_afrag(A, lda, m0, kk, lane);
    v16bf a1 = load_afrag(A, lda, m0 + 16, kk, lane);
#pragma unroll
    for (int ni = 0; ni < 4; ++ni) {
      v16bf b = load_bfrag_nk(Bt, ldb, n0 + ni * 16, kk, lane);
      c0[ni] = wmma_bf16(a0, b, c0[ni]);
      c1[ni] = wmma_bf16(a1, b, c1[ni]);
    }
  }

#pragma unroll
  for (int mi = 0; mi < 2; ++mi) {
#pragma unroll
    for (int ni = 0; ni < 4; ++ni) {
      const v8f& c = mi ? c1[ni] : c0[ni];
      int col = n0 + ni * 16 + (lane & 15);
      float bv_ = bias ? bias[col] : 0.f;
#pragma unroll
      for (int j = 0; j < 8; ++j) {
        int row = m0 + mi * 16 + j + ((lane & 16) ? 8 : 0);
        float v = c[j] + bv_;
        if (flags & GF_RELU) v = v > 0.f ? v : 0.f;
        if (flags & GF_RESID) v += resid[(size_t)row * N + col];
        if (flags & GF_F32)  Cf[(size_t)row * N + col] = v;
        if (flags & GF_BF16) Cb[(size_t)row * N + col] = (__bf16)v;
        if (flags & GF_QKV) {
          int sec = col >> 8, nl = col & 255, head = nl >> 5, d = nl & 31;
          int bi = row >> 9, l = row & 511;
          size_t o = (((size_t)(bi * NH + head)) * LL + l) * DK + d;
          __bf16 bw = (__bf16)v;
          if (sec == 0) qd[o] = bw; else if (sec == 1) kd[o] = bw; else vd[o] = bw;
        }
      }
    }
  }
}

// ----------------------- fused relation-aware attention --------------------
// One wave per (b,h, 16-row q tile). Scores for the full 512-wide row kept in
// LDS; relation bias gathered from qrelb; softmax + 37-bin prel histogram via
// ds_add_f32; P@V by WMMA with P converted bf16 from LDS; prel@rel_v epilogue.
__global__ void attn_kernel(const __bf16* __restrict__ q, const __bf16* __restrict__ k,
                            const __bf16* __restrict__ v, const float* __restrict__ qrelb,
                            const int* __restrict__ rel, const float* __restrict__ relv,
                            __bf16* __restrict__ attnO) {
  __shared__ float sc[16][LL];            // 32 KB
  __shared__ unsigned char rid[16][LL];   // 8 KB
  __shared__ float prel[16][RPAD];        // 2.5 KB
  int lane = threadIdx.x;
  int bh = blockIdx.x >> 5;               // b*NH + h
  int qt = blockIdx.x & 31;
  int b = bh >> 3, h = bh & 7;
  int q0 = qt << 4;
  const __bf16* qbase = q + (size_t)bh * LL * DK;
  const __bf16* kbase = k + (size_t)bh * LL * DK;
  const __bf16* vbase = v + (size_t)bh * LL * DK;

  for (int i = lane; i < 16 * RPAD; i += 32) (&prel[0][0])[i] = 0.f;

  // --- scores = (Q K^T)/sqrt(dk) + qrel[rel] ; dk==32 -> one WMMA per tile
  v16bf aq = load_afrag(qbase, DK, q0, 0, lane);
  for (int kt = 0; kt < LL / 16; ++kt) {
    v16bf bk = load_bfrag_nk(kbase, DK, kt * 16, 0, lane);
    v8f c = {};
    c = wmma_bf16(aq, bk, c);
    int col = kt * 16 + (lane & 15);
#pragma unroll
    for (int j = 0; j < 8; ++j) {
      int m = j + ((lane & 16) ? 8 : 0);
      int qg = q0 + m;
      int r = rel[((size_t)(b * LL + qg)) * LL + col];
      sc[m][col] = c[j] * RSQRT_DK + qrelb[(size_t)(bh * LL + qg) * RPAD + r];
      rid[m][col] = (unsigned char)r;
    }
  }
  __syncthreads();

  // --- row softmax + prel histogram
  for (int m = 0; m < 16; ++m) {
    float mx = -1e30f;
    for (int i = lane; i < LL; i += 32) mx = fmaxf(mx, sc[m][i]);
#pragma unroll
    for (int s = 16; s; s >>= 1) mx = fmaxf(mx, __shfl_xor(mx, s, 32));
    float sum = 0.f;
    for (int i = lane; i < LL; i += 32) {
      float e = __expf(sc[m][i] - mx);
      sc[m][i] = e;
      sum += e;
    }
#pragma unroll
    for (int s = 16; s; s >>= 1) sum += __shfl_xor(sum, s, 32);
    float inv = 1.f / sum;
    for (int i = lane; i < LL; i += 32) {
      float p = sc[m][i] * inv;
      sc[m][i] = p;
      atomicAdd(&prel[m][rid[m][i]], p);   // ds_add_f32
    }
  }
  __syncthreads();

  // --- O = P @ V (two 16x16 tiles over d), P read bf16 from LDS
  v8f c0 = {}, c1 = {};
  for (int kt = 0; kt < LL / 32; ++kt) {
    int m = lane & 15;
    int kb = kt * 32 + ((lane & 16) ? 16 : 0);
    v16bf a;
#pragma unroll
    for (int e = 0; e < 8; ++e) a[e] = (__bf16)sc[m][kb + e];
#pragma unroll
    for (int e = 0; e < 8; ++e) a[8 + e] = (__bf16)sc[m][kb + 8 + e];
    v16bf b0 = load_bfrag_kn(vbase, DK, kt * 32, 0, lane);
    v16bf b1 = load_bfrag_kn(vbase, DK, kt * 32, 16, lane);
    c0 = wmma_bf16(a, b0, c0);
    c1 = wmma_bf16(a, b1, c1);
  }

  // --- epilogue: + prel @ rel_v, store concat-head bf16 [4096][256]
  int d = lane & 15;
#pragma unroll
  for (int j = 0; j < 8; ++j) {
    int m = j + ((lane & 16) ? 8 : 0);
    float a0 = c0[j], a1 = c1[j];
    for (int r = 0; r < NREL; ++r) {
      float p = prel[m][r];
      a0 += p * relv[r * DK + d];
      a1 += p * relv[r * DK + 16 + d];
    }
    size_t orow = (size_t)(b * LL + q0 + m) * HH + h * DK;
    attnO[orow + d] = (__bf16)a0;
    attnO[orow + 16 + d] = (__bf16)a1;
  }
}

// -------------------- align (pointer) attention, single head ---------------
// One wave per (b, 16-row q tile); scores over kvlen (192 or 64) columns.
__global__ void align_attn_k(const __bf16* __restrict__ aq, const __bf16* __restrict__ ak,
                             const float* __restrict__ aqrel, const int* __restrict__ rel,
                             int kvbase, int kvlen, float* __restrict__ out) {
  __shared__ float sc[16][192];
  int lane = threadIdx.x;
  int b = blockIdx.x >> 5;
  int q0 = (blockIdx.x & 31) << 4;
  int ntiles = kvlen >> 4;

  for (int nt = 0; nt < ntiles; ++nt) {
    v8f c = {};
    for (int kk = 0; kk < HH; kk += 32) {
      v16bf a = load_afrag(aq, HH, b * LL + q0, kk, lane);
      v16bf bb = load_bfrag_nk(ak, HH, b * LL + kvbase + nt * 16, kk, lane);
      c = wmma_bf16(a, bb, c);
    }
    int col = nt * 16 + (lane & 15);
#pragma unroll
    for (int j = 0; j < 8; ++j) {
      int m = j + ((lane & 16) ? 8 : 0);
      int qg = q0 + m;
      int r = rel[((size_t)(b * LL + qg)) * LL + kvbase + col];
      sc[m][col] = c[j] * ALIGN_SCALE + aqrel[(size_t)(b * LL + qg) * RPAD + r];
    }
  }
  __syncthreads();

  for (int m = 0; m < 16; ++m) {
    float mx = -1e30f;
    for (int i = lane; i < kvlen; i += 32) mx = fmaxf(mx, sc[m][i]);
#pragma unroll
    for (int s = 16; s; s >>= 1) mx = fmaxf(mx, __shfl_xor(mx, s, 32));
    float sum = 0.f;
    for (int i = lane; i < kvlen; i += 32) {
      float e = __expf(sc[m][i] - mx);
      sc[m][i] = e;
      sum += e;
    }
#pragma unroll
    for (int s = 16; s; s >>= 1) sum += __shfl_xor(sum, s, 32);
    float inv = 1.f / sum;
    int qg = q0 + m;
    for (int i = lane; i < kvlen; i += 32)
      out[((size_t)(b * LL + qg)) * kvlen + i] = sc[m][i] * inv;
  }
}

// ------------------------------- host side ---------------------------------

extern "C" void kernel_launch(void* const* d_in, const int* in_sizes, int n_in,
                              void* d_out, int out_size, void* d_ws, size_t ws_size,
                              hipStream_t stream) {
  // input order: enc, relations, mask, c_base, t_base, then params in dict order
  const float* enc   = (const float*)d_in[0];
  const int*   rel   = (const int*)d_in[1];
  const float* Wq    = (const float*)d_in[5];
  const float* bq    = (const float*)d_in[6];
  const float* Wk    = (const float*)d_in[7];
  const float* bk    = (const float*)d_in[8];
  const float* Wv    = (const float*)d_in[9];
  const float* bv    = (const float*)d_in[10];
  const float* Wo    = (const float*)d_in[11];
  const float* bo    = (const float*)d_in[12];
  const float* relk  = (const float*)d_in[13];
  const float* relv  = (const float*)d_in[14];
  const float* ln1g  = (const float*)d_in[15];
  const float* ln1b  = (const float*)d_in[16];
  const float* ln2g  = (const float*)d_in[17];
  const float* ln2b  = (const float*)d_in[18];
  const float* W1    = (const float*)d_in[19];
  const float* b1    = (const float*)d_in[20];
  const float* W2    = (const float*)d_in[21];
  const float* b2    = (const float*)d_in[22];
  const float* lnfg  = (const float*)d_in[23];
  const float* lnfb  = (const float*)d_in[24];
  const float* aWq   = (const float*)d_in[25];
  const float* abq   = (const float*)d_in[26];
  const float* aWk   = (const float*)d_in[27];
  const float* abk   = (const float*)d_in[28];
  const float* arelk = (const float*)d_in[29];

  // workspace carve-up
  char* w = (char*)d_ws;
  size_t off = 0;
  auto alloc = [&](size_t bytes) -> void* {
    void* p = w + off;
    off += (bytes + 255) & ~(size_t)255;
    return p;
  };
  float*  x      = (float*)  alloc((size_t)ROWS * HH * 4);
  __bf16* hb     = (__bf16*) alloc((size_t)ROWS * HH * 2);
  __bf16* qb     = (__bf16*) alloc((size_t)BHROW * DK * 2);
  __bf16* kbuf   = (__bf16*) alloc((size_t)BHROW * DK * 2);
  __bf16* vbuf   = (__bf16*) alloc((size_t)BHROW * DK * 2);
  float*  qrelb  = (float*)  alloc((size_t)BHROW * RPAD * 4);
  __bf16* attnO  = (__bf16*) alloc((size_t)ROWS * HH * 2);
  __bf16* ffh    = (__bf16*) alloc((size_t)ROWS * FFD * 2);
  __bf16* aqb    = (__bf16*) alloc((size_t)ROWS * HH * 2);
  __bf16* akb    = (__bf16*) alloc((size_t)ROWS * HH * 2);
  float*  aqrel  = (float*)  alloc((size_t)ROWS * RPAD * 4);
  __bf16* wqkv   = (__bf16*) alloc((size_t)NL * 768 * 256 * 2);
  __bf16* wo_t   = (__bf16*) alloc((size_t)NL * 256 * 256 * 2);
  __bf16* w1_t   = (__bf16*) alloc((size_t)NL * 1024 * 256 * 2);
  __bf16* w2_t   = (__bf16*) alloc((size_t)NL * 256 * 1024 * 2);
  __bf16* awq_t  = (__bf16*) alloc((size_t)256 * 256 * 2);
  __bf16* awk_t  = (__bf16*) alloc((size_t)256 * 256 * 2);
  float*  qkvbias= (float*)  alloc((size_t)NL * 768 * 4);

  float* outx  = (float*)d_out;                       // [8,512,256]
  float* outc  = outx + (size_t)ROWS * HH;            // [8,512,192]
  float* outt  = outc + (size_t)ROWS * 192;           // [8,512,64]

  // --- weight prep (every call; deterministic) -----------------------------
  dim3 t256(256);
  transpose_bf16_k<<<dim3(256, NL), t256, 0, stream>>>(Wq, wqkv,          256, 256, 65536, 196608);
  transpose_bf16_k<<<dim3(256, NL), t256, 0, stream>>>(Wk, wqkv + 65536,  256, 256, 65536, 196608);
  transpose_bf16_k<<<dim3(256, NL), t256, 0, stream>>>(Wv, wqkv + 131072, 256, 256, 65536, 196608);
  transpose_bf16_k<<<dim3(256, NL), t256, 0, stream>>>(Wo, wo_t, 256, 256, 65536, 65536);
  transpose_bf16_k<<<dim3(1024, NL), t256, 0, stream>>>(W1, w1_t, 256, 1024, 262144, 262144);
  transpose_bf16_k<<<dim3(1024, NL), t256, 0, stream>>>(W2, w2_t, 1024, 256, 262144, 262144);
  transpose_bf16_k<<<dim3(256, 1), t256, 0, stream>>>(aWq, awq_t, 256, 256, 0, 0);
  transpose_bf16_k<<<dim3(256, 1), t256, 0, stream>>>(aWk, awk_t, 256, 256, 0, 0);
  qkv_bias_concat<<<24, 256, 0, stream>>>(bq, bk, bv, qkvbias);

  hipMemcpyAsync(x, enc, (size_t)ROWS * HH * 4, hipMemcpyDeviceToDevice, stream);

  // wave-tile grids: (M/32)*(N/64) waves, 4 waves per 128-thread block
  const int gQKV = (128 * 12) / 4;   // M=4096, N=768
  const int gH   = (128 * 4) / 4;    // M=4096, N=256
  const int gFF  = (128 * 16) / 4;   // M=4096, N=1024

  // --- layers --------------------------------------------------------------
  for (int i = 0; i < NL; ++i) {
    // LN1 -> hb (bf16)
    ln_kernel<<<ROWS, HH, 0, stream>>>(x, ln1g + i * HH, ln1b + i * HH, nullptr, hb);
    // QKV: [4096,256] x [256,768]  -> per-head bf16 q/k/v
    gemm_wmma_k<<<gQKV, 128, 0, stream>>>(
        hb, HH, wqkv + (size_t)i * 768 * 256, 256, qkvbias + i * 768, nullptr,
        nullptr, nullptr, ROWS, 768, HH, GF_QKV, qb, kbuf, vbuf);
    // qrel bias table [32768][37]
    qrel_kernel<<<(BHROW * NREL + 255) / 256, 256, 0, stream>>>(
        qb, relk + (size_t)i * NREL * DK, qrelb, BHROW, DK, DK, RSQRT_DK);
    // fused attention (scores+softmax+prel+PV+rel_v) -> attnO bf16
    attn_kernel<<<BB * NH * (LL / 16), 32, 0, stream>>>(
        qb, kbuf, vbuf, qrelb, rel, relv + (size_t)i * NREL * DK, attnO);
    // x += attnO @ Wo + bo
    gemm_wmma_k<<<gH, 128, 0, stream>>>(
        attnO, HH, wo_t + (size_t)i * 256 * 256, 256, bo + i * HH, x,
        x, nullptr, ROWS, HH, HH, GF_RESID | GF_F32, nullptr, nullptr, nullptr);
    // LN2 -> hb
    ln_kernel<<<ROWS, HH, 0, stream>>>(x, ln2g + i * HH, ln2b + i * HH, nullptr, hb);
    // ffh = relu(hb @ W1 + b1) (bf16)
    gemm_wmma_k<<<gFF, 128, 0, stream>>>(
        hb, HH, w1_t + (size_t)i * 1024 * 256, 256, b1 + i * FFD, nullptr,
        nullptr, ffh, ROWS, FFD, HH, GF_RELU | GF_BF16, nullptr, nullptr, nullptr);
    // x += ffh @ W2 + b2
    gemm_wmma_k<<<gH, 128, 0, stream>>>(
        ffh, FFD, w2_t + (size_t)i * 256 * 1024, 1024, b2 + i * HH, x,
        x, nullptr, ROWS, HH, FFD, GF_RESID | GF_F32, nullptr, nullptr, nullptr);
  }

  // --- final LN (writes fp32 output + bf16 copy for align attention) -------
  ln_kernel<<<ROWS, HH, 0, stream>>>(x, lnfg, lnfb, outx, hb);

  // align projections
  gemm_wmma_k<<<gH, 128, 0, stream>>>(
      hb, HH, awq_t, 256, abq, nullptr, nullptr, aqb, ROWS, HH, HH, GF_BF16,
      nullptr, nullptr, nullptr);
  gemm_wmma_k<<<gH, 128, 0, stream>>>(
      hb, HH, awk_t, 256, abk, nullptr, nullptr, akb, ROWS, HH, HH, GF_BF16,
      nullptr, nullptr, nullptr);
  // align relation bias table [4096][37]
  qrel_kernel<<<(ROWS * NREL + 255) / 256, 256, 0, stream>>>(
      aqb, arelk, aqrel, ROWS, HH, HH, ALIGN_SCALE);
  // m2c over columns [256,448), m2t over tables [448,512)
  align_attn_k<<<BB * (LL / 16), 32, 0, stream>>>(aqb, akb, aqrel, rel, 256, 192, outc);
  align_attn_k<<<BB * (LL / 16), 32, 0, stream>>>(aqb, akb, aqrel, rel, 448, 64, outt);
}
